// GQASelfAttention_62878321213691
// MI455X (gfx1250) — matrix operand order repaired and checked by reading, not compiled, not run on hardware
//
#include <hip/hip_runtime.h>
#include <stdint.h>

#define BB 2
#define SS 2048
#define DD 1024
#define NH 16
#define NKV 4
#define HD 64
#define KVD (NKV * HD) /* 256 */

typedef __attribute__((ext_vector_type(16))) __bf16 v16bf;
typedef __attribute__((ext_vector_type(8)))  float  v8f;

// ---------------- bf16 helpers ----------------
__device__ __forceinline__ unsigned short f2bf(float f) {
  unsigned int u = __float_as_uint(f);
  u += 0x7FFFu + ((u >> 16) & 1u);   // round to nearest even
  return (unsigned short)(u >> 16);
}
__device__ __forceinline__ float bf2f(unsigned short s) {
  return __uint_as_float(((unsigned int)s) << 16);
}

union FragB16 { v16bf v; unsigned int u[8]; unsigned short s[16]; };

__device__ __forceinline__ v8f zero_v8f() {
  v8f z = {0.f, 0.f, 0.f, 0.f, 0.f, 0.f, 0.f, 0.f};
  return z;
}

// CDNA5 async copy: global -> LDS, 16 bytes per lane, tracked by ASYNCcnt.
__device__ __forceinline__ void async_b128(unsigned lds_addr, const void* gaddr) {
  asm volatile("global_load_async_to_lds_b128 %0, %1, off"
               :: "v"(lds_addr), "v"(gaddr) : "memory");
}
__device__ __forceinline__ void wait_async0() {
  asm volatile("s_wait_asynccnt 0x0" ::: "memory");
}
__device__ __forceinline__ void wait_ds0() {
  asm volatile("s_wait_dscnt 0x0" ::: "memory");
}
__device__ __forceinline__ unsigned lds_addr_of(const void* p) {
  return (unsigned)(uintptr_t)p;  // LDS aperture: hw uses addr[31:0]
}

// A fragment (16x32 bf16, MxK): element (m,k) at p[m*stride + k].
__device__ __forceinline__ v16bf load_frag_a(const unsigned short* p, int stride) {
  int lane = threadIdx.x & 31;
  int m = lane & 15, half = lane >> 4;
  const unsigned short* row = p + m * stride + half * 8;
  FragB16 f;
#pragma unroll
  for (int j = 0; j < 4; ++j) {
    f.u[j]     = *(const unsigned int*)(row + 2 * j);
    f.u[4 + j] = *(const unsigned int*)(row + 16 + 2 * j);
  }
  return f.v;
}

// B fragment (32x16 bf16, KxN): element (k,n) at p[n*stride + k] (k contiguous)
__device__ __forceinline__ v16bf load_frag_b(const unsigned short* p, int stride) {
  int lane = threadIdx.x & 31;
  int n = lane & 15, kh = lane >> 4;
  const unsigned short* col = p + n * stride + kh * 8;
  FragB16 f;
#pragma unroll
  for (int j = 0; j < 4; ++j) {
    f.u[j]     = *(const unsigned int*)(col + 2 * j);
    f.u[4 + j] = *(const unsigned int*)(col + 16 + 2 * j);
  }
  return f.v;
}

__device__ __forceinline__ void store_out(float* p, float v) { *p = v; }
__device__ __forceinline__ void store_out(unsigned short* p, float v) { *p = f2bf(v); }

// ---------------- f32 -> bf16 conversion (plain + transposing) ----------------
__global__ void cvt_bf16_kernel(const float* __restrict__ in,
                                unsigned short* __restrict__ out, int n) {
  int i = blockIdx.x * blockDim.x + threadIdx.x;
  if (i < n) out[i] = f2bf(in[i]);
}

// in: [K][N] row-major f32  ->  out: [N][K] row-major bf16
__global__ void cvt_bf16_t_kernel(const float* __restrict__ in,
                                  unsigned short* __restrict__ out, int K, int N) {
  int i = blockIdx.x * blockDim.x + threadIdx.x;
  if (i >= K * N) return;
  int k = i / N, n = i % N;
  out[(size_t)n * K + k] = f2bf(in[i]);
}

// ---------------- tiled WMMA GEMM: C[MxN] = A[MxK] * Wt[NxK]^T ----------------
// 256 threads = 8 waves; block tile 128x128; wave tile 32x64 (2x4 WMMA tiles).
// Double-buffered LDS fed by CDNA5 async global->LDS copies.
template <typename OutT>
__global__ __launch_bounds__(256) void gemm_bf16_wmma(
    const unsigned short* __restrict__ A,   // [M][K]
    const unsigned short* __restrict__ Wt,  // [N][K] (pre-transposed)
    OutT* __restrict__ C, int M, int N, int K) {
  __shared__ __align__(16) unsigned short As[2][128 * 32];  // [m][k]
  __shared__ __align__(16) unsigned short Bs[2][128 * 32];  // [n][k]

  const int bx = blockIdx.x;  // N tile
  const int by = blockIdx.y;  // M tile
  const int tid = threadIdx.x;
  const int wave = tid >> 5;
  const int wm = wave & 3;   // 4 row-groups of 32
  const int wn = wave >> 2;  // 2 col-groups of 64

  // per-thread staging slot: 16 contiguous bf16 (32B) of one tile row
  const int r = tid & 127, seg = (tid >> 7) * 16;
  const unsigned short* ga = A + (size_t)(by * 128 + r) * K + seg;
  const unsigned short* gb = Wt + (size_t)(bx * 128 + r) * K + seg;
  const unsigned la0 = lds_addr_of(&As[0][r * 32 + seg]);
  const unsigned lb0 = lds_addr_of(&Bs[0][r * 32 + seg]);
  const unsigned bufstep = 128 * 32 * 2;  // bytes between buffers

  // prologue: stream tile k0=0 into buffer 0
  async_b128(la0, ga);      async_b128(la0 + 16, ga + 8);
  async_b128(lb0, gb);      async_b128(lb0 + 16, gb + 8);

  v8f acc[2][4];
#pragma unroll
  for (int i = 0; i < 2; ++i)
#pragma unroll
    for (int j = 0; j < 4; ++j) acc[i][j] = zero_v8f();

  int buf = 0;
  for (int k0 = 0; k0 < K; k0 += 32, buf ^= 1) {
    wait_async0();       // this wave's async copies into `buf` have landed
    __syncthreads();     // everyone's copies landed; prior reads of buf^1 done

    if (k0 + 32 < K) {   // stream next tile into the other buffer
      const unsigned short* ga2 = ga + k0 + 32;
      const unsigned short* gb2 = gb + k0 + 32;
      unsigned la = la0 + (unsigned)(buf ^ 1) * bufstep;
      unsigned lb = lb0 + (unsigned)(buf ^ 1) * bufstep;
      async_b128(la, ga2);      async_b128(la + 16, ga2 + 8);
      async_b128(lb, gb2);      async_b128(lb + 16, gb2 + 8);
    }

    v16bf afrag[2];
#pragma unroll
    for (int i = 0; i < 2; ++i)
      afrag[i] = load_frag_a(&As[buf][(wm * 32 + i * 16) * 32], 32);
#pragma unroll
    for (int j = 0; j < 4; ++j) {
      v16bf bfrag = load_frag_b(&Bs[buf][(wn * 64 + j * 16) * 32], 32);
#pragma unroll
      for (int i = 0; i < 2; ++i)
        acc[i][j] = __builtin_amdgcn_wmma_f32_16x16x32_bf16(
            false, afrag[i], false, bfrag, (short)0, acc[i][j], false, false);
    }
  }

  // C fragment store: row = vgpr + 8*(lane>>4), col = lane&15
  const int lane = tid & 31;
  const int n16 = lane & 15, half = lane >> 4;
#pragma unroll
  for (int i = 0; i < 2; ++i)
#pragma unroll
    for (int j = 0; j < 4; ++j) {
      int row0 = by * 128 + wm * 32 + i * 16 + half * 8;
      int col = bx * 128 + wn * 64 + j * 16 + n16;
#pragma unroll
      for (int rr = 0; rr < 8; ++rr)
        store_out(C + (size_t)(row0 + rr) * N + col, acc[i][j][rr]);
    }
}

// ---------------- RoPE (rotate-half, full head dim) in place ----------------
__global__ void rope_kernel(unsigned short* __restrict__ t, int nheads) {
  int idx = blockIdx.x * blockDim.x + threadIdx.x;
  int total = BB * SS * nheads * 32;
  if (idx >= total) return;
  int i = idx & 31;
  int h = (idx >> 5) % nheads;
  int s = ((idx >> 5) / nheads) % SS;
  int b = idx / (32 * nheads * SS);
  float ang = (float)s * __expf(-(float)i * (9.210340371976184f / 32.0f));
  float cs = cosf(ang), sn = sinf(ang);
  size_t base = (((size_t)b * SS + s) * nheads + h) * HD;
  float a0 = bf2f(t[base + i]);
  float a1 = bf2f(t[base + 32 + i]);
  t[base + i]      = f2bf(a0 * cs - a1 * sn);
  t[base + 32 + i] = f2bf(a0 * sn + a1 * cs);
}

// ---------------- flash attention (causal, GQA kv = h % 4) ----------------
// grid: (S/64, NH, B); 128 threads = 4 waves, each wave owns 16 query rows.
__global__ __launch_bounds__(128) void flash_attn_kernel(
    const unsigned short* __restrict__ Q, const unsigned short* __restrict__ K,
    const unsigned short* __restrict__ V, unsigned short* __restrict__ O) {
  __shared__ __align__(16) unsigned short Ks[64 * 64];     // [key][dim]
  __shared__ __align__(16) unsigned short Vs[64 * 64];     // [dim][key]
  __shared__ __align__(16) unsigned short Ps[4][16 * 64];  // per-wave P [row][key]

  const int qb = blockIdx.x, h = blockIdx.y, b = blockIdx.z;
  const int kv = h & (NKV - 1);  // reference reshape => kv head = h % 4
  const int tid = threadIdx.x, wave = tid >> 5, lane = tid & 31;
  const int n16 = lane & 15, half = lane >> 4;
  const int q0 = qb * 64 + wave * 16;

  // Q fragments: 16 rows x 64 dims = two 16x32 A fragments, loaded once.
  v16bf qfrag[2];
  const unsigned short* qbase = Q + ((size_t)(b * SS + q0)) * DD + h * HD;
#pragma unroll
  for (int kc = 0; kc < 2; ++kc) qfrag[kc] = load_frag_a(qbase + kc * 32, DD);

  // per-thread K staging slot: row kr, 32 contiguous bf16 (64B) = 4 async b128
  const int kr = tid >> 1, kseg = (tid & 1) * 32;
  const unsigned lk = lds_addr_of(&Ks[kr * 64 + kseg]);

  float mrow[8], lrow[8];
#pragma unroll
  for (int rr = 0; rr < 8; ++rr) { mrow[rr] = -1e30f; lrow[rr] = 0.f; }
  v8f o[4];
#pragma unroll
  for (int n = 0; n < 4; ++n) o[n] = zero_v8f();

  const float scale = 0.125f;  // 1/sqrt(64)

  for (int jb = 0; jb <= qb; ++jb) {
    __syncthreads();  // previous iteration done reading Ks/Vs
    {  // K tile [key][dim] via CDNA5 async global->LDS
      const unsigned short* gk =
          K + ((size_t)(b * SS + jb * 64 + kr)) * KVD + kv * HD + kseg;
      async_b128(lk,      gk);
      async_b128(lk + 16, gk + 8);
      async_b128(lk + 32, gk + 16);
      async_b128(lk + 48, gk + 24);
    }
    {  // V tile transposed -> Vs[dim][key] (overlaps the K async copies)
      int key = tid >> 1, dseg = (tid & 1) * 32;
      const unsigned short* g =
          V + ((size_t)(b * SS + jb * 64 + key)) * KVD + kv * HD + dseg;
      union { uint4 q[4]; unsigned short s[32]; } t;
      t.q[0] = ((const uint4*)g)[0]; t.q[1] = ((const uint4*)g)[1];
      t.q[2] = ((const uint4*)g)[2]; t.q[3] = ((const uint4*)g)[3];
#pragma unroll
      for (int i = 0; i < 32; ++i) Vs[(dseg + i) * 64 + key] = t.s[i];
    }
    wait_async0();
    __syncthreads();

    // S = Q K^T : 4 key-subtiles x 2 k-chunks
    v8f sc[4];
#pragma unroll
    for (int n = 0; n < 4; ++n) {
      sc[n] = zero_v8f();
#pragma unroll
      for (int kc = 0; kc < 2; ++kc) {
        v16bf bfrag = load_frag_b(Ks + (n * 16) * 64 + kc * 32, 64);
        sc[n] = __builtin_amdgcn_wmma_f32_16x16x32_bf16(
            false, qfrag[kc], false, bfrag, (short)0, sc[n], false, false);
      }
    }

    // Online softmax over the C-fragment layout.
#pragma unroll
    for (int rr = 0; rr < 8; ++rr) {
      int qrow = q0 + half * 8 + rr;
      float mx = -1e30f;
#pragma unroll
      for (int n = 0; n < 4; ++n) {
        int key = jb * 64 + n * 16 + n16;
        float s = sc[n][rr] * scale;
        s = (key <= qrow) ? s : -1e30f;  // causal mask
        sc[n][rr] = s;
        mx = fmaxf(mx, s);
      }
#pragma unroll
      for (int off = 8; off > 0; off >>= 1)
        mx = fmaxf(mx, __shfl_xor(mx, off, 32));  // reduce across lane half
      float mnew = fmaxf(mrow[rr], mx);
      float corr = __expf(mrow[rr] - mnew);
      mrow[rr] = mnew;
      float rs = 0.f;
#pragma unroll
      for (int n = 0; n < 4; ++n) {
        float p = __expf(sc[n][rr] - mnew);
        rs += p;
        Ps[wave][(half * 8 + rr) * 64 + n * 16 + n16] = f2bf(p);
      }
#pragma unroll
      for (int off = 8; off > 0; off >>= 1) rs += __shfl_xor(rs, off, 32);
      lrow[rr] = lrow[rr] * corr + rs;
#pragma unroll
      for (int n = 0; n < 4; ++n) o[n][rr] *= corr;
    }
    // in-wave LDS RAW: P written above, re-read below as A fragments
    wait_ds0();

    // O += P V
#pragma unroll
    for (int kc = 0; kc < 2; ++kc) {
      v16bf pfrag = load_frag_a(&Ps[wave][kc * 32], 64);
#pragma unroll
      for (int n = 0; n < 4; ++n) {
        v16bf vfrag = load_frag_b(Vs + (n * 16) * 64 + kc * 32, 64);
        o[n] = __builtin_amdgcn_wmma_f32_16x16x32_bf16(
            false, pfrag, false, vfrag, (short)0, o[n], false, false);
      }
    }
  }

  // epilogue: O / l, write bf16 [B,S,H,HD] (== [B,S,D])
#pragma unroll
  for (int n = 0; n < 4; ++n)
#pragma unroll
    for (int rr = 0; rr < 8; ++rr) {
      int row = q0 + half * 8 + rr;
      float val = o[n][rr] / lrow[rr];
      O[((size_t)(b * SS + row)) * DD + h * HD + n * 16 + n16] = f2bf(val);
    }
}

// ---------------- host launcher ----------------
extern "C" void kernel_launch(void* const* d_in, const int* in_sizes, int n_in,
                              void* d_out, int out_size, void* d_ws, size_t ws_size,
                              hipStream_t stream) {
  const float* x  = (const float*)d_in[0];
  const float* wq = (const float*)d_in[1];
  const float* wk = (const float*)d_in[2];
  const float* wv = (const float*)d_in[3];
  const float* wo = (const float*)d_in[4];
  float* out = (float*)d_out;

  char* ws = (char*)d_ws;
  size_t off = 0;
  auto carve = [&](size_t bytes) -> unsigned short* {
    size_t a = (off + 255) & ~(size_t)255;
    off = a + bytes;
    return (unsigned short*)(ws + a);
  };

  const int M = BB * SS;  // 4096
  unsigned short* xb  = carve((size_t)M * DD * 2);   // bf16 x (reused as attn out)
  unsigned short* wqb = carve((size_t)DD * DD * 2);  // transposed [N][K]
  unsigned short* wkb = carve((size_t)DD * KVD * 2);
  unsigned short* wvb = carve((size_t)DD * KVD * 2);
  unsigned short* wob = carve((size_t)DD * DD * 2);
  unsigned short* qb  = carve((size_t)M * DD * 2);
  unsigned short* kb  = carve((size_t)M * KVD * 2);
  unsigned short* vb  = carve((size_t)M * KVD * 2);
  unsigned short* ob  = xb;  // safe: x no longer needed after QKV projections

  // x -> bf16; weights -> transposed bf16 [N][K]
  cvt_bf16_kernel<<<(M * DD) / 256, 256, 0, stream>>>(x, xb, M * DD);
  cvt_bf16_t_kernel<<<(DD * DD) / 256, 256, 0, stream>>>(wq, wqb, DD, DD);
  cvt_bf16_t_kernel<<<(DD * KVD) / 256, 256, 0, stream>>>(wk, wkb, DD, KVD);
  cvt_bf16_t_kernel<<<(DD * KVD) / 256, 256, 0, stream>>>(wv, wvb, DD, KVD);
  cvt_bf16_t_kernel<<<(DD * DD) / 256, 256, 0, stream>>>(wo, wob, DD, DD);

  // projections
  gemm_bf16_wmma<unsigned short>
      <<<dim3(DD / 128, M / 128), 256, 0, stream>>>(xb, wqb, qb, M, DD, DD);
  gemm_bf16_wmma<unsigned short>
      <<<dim3(KVD / 128, M / 128), 256, 0, stream>>>(xb, wkb, kb, M, KVD, DD);
  gemm_bf16_wmma<unsigned short>
      <<<dim3(KVD / 128, M / 128), 256, 0, stream>>>(xb, wvb, vb, M, KVD, DD);

  // RoPE in place
  rope_kernel<<<(BB * SS * NH * 32) / 256, 256, 0, stream>>>(qb, NH);
  rope_kernel<<<(BB * SS * NKV * 32) / 256, 256, 0, stream>>>(kb, NKV);

  // flash attention
  flash_attn_kernel<<<dim3(SS / 64, NH, BB), 128, 0, stream>>>(qb, kb, vb, ob);

  // output projection -> f32
  gemm_bf16_wmma<float>
      <<<dim3(DD / 128, M / 128), 256, 0, stream>>>(ob, wob, out, M, DD, DD);
}
